// TGN_58119497450033
// MI455X (gfx1250) — matrix operand order repaired and checked by reading, not compiled, not run on hardware
//
#include <hip/hip_runtime.h>
#include <hip/hip_bf16.h>

typedef __bf16 v16bf __attribute__((ext_vector_type(16)));
typedef float  v8f   __attribute__((ext_vector_type(8)));

#define N_NODES   200000
#define N_EVENTS  100000
#define MEM_DIM   100
#define TIME_DIM  100
#define MSG_DIM   172
#define SCAT_W    372            // scattered cols: mem_other(100)+raw(172)+tenc(100)
#define K1        472            // full aggr width
#define KT1       15             // 480/32 k-tiles for gi GEMM
#define KT2       4              // 128/32 k-tiles for gh GEMM
#define NOUT      300            // 3*MEM_DIM gate width
#define NTILES    19             // ceil(300/16)
#define GPAD      (NTILES * 16 + 4)   // padded gi/gh row stride (bank-conflict free)
#define MTILE     32             // nodes per block (2 x 16-row WMMA subtiles)
#define NBLOCKS   (N_NODES / MTILE)   // 6250 exactly

// ---- round-to-nearest-even f32 -> bf16 (no reliance on hw cvt) ----
__device__ __forceinline__ __bf16 f2bf(float f) {
    union { float f; unsigned u; } v; v.f = f;
    unsigned r = v.u + 0x7fffu + ((v.u >> 16) & 1u);
    unsigned short h = (unsigned short)(r >> 16);
    return __builtin_bit_cast(__bf16, h);
}

__device__ __forceinline__ float sigmoidf_(float x) {
    return 1.0f / (1.0f + __expf(-x));
}

// Native CDNA5 fp32 atomic add (non-returning -> STOREcnt path, no CAS loop).
__device__ __forceinline__ void atomic_add_f32_dev(float* p, float v) {
    asm volatile("global_atomic_add_f32 %0, %1, off scope:SCOPE_DEV"
                 :: "v"(p), "v"(v) : "memory");
}

// =====================================================================
// Kernel 1: pack w_ih / w_hh into bf16 WMMA B-fragment layout.
// B (32x16 K x N) per lane: n = lane&15, k = ktile*32 + 16*(lane>>4) + e,
// 16 contiguous bf16 per lane => one 32B load per fragment at run time.
// =====================================================================
__global__ __launch_bounds__(256) void pack_weights(
        const float* __restrict__ w_ih, const float* __restrict__ w_hh,
        __bf16* __restrict__ pWih, __bf16* __restrict__ pWhh) {
    const int wihN = KT1 * NTILES * 512;
    const int whhN = KT2 * NTILES * 512;
    int i = blockIdx.x * 256 + threadIdx.x;
    if (i < wihN) {
        int kt = i / (NTILES * 512);
        int r  = i % (NTILES * 512);
        int nt = r / 512;
        int q  = r % 512;
        int lane = q >> 4, e = q & 15;
        int n = nt * 16 + (lane & 15);
        int k = kt * 32 + ((lane >> 4) << 4) + e;
        float v = (n < NOUT && k < K1) ? w_ih[n * K1 + k] : 0.0f;
        pWih[i] = f2bf(v);
    } else if (i < wihN + whhN) {
        int j  = i - wihN;
        int kt = j / (NTILES * 512);
        int r  = j % (NTILES * 512);
        int nt = r / 512;
        int q  = r % 512;
        int lane = q >> 4, e = q & 15;
        int n = nt * 16 + (lane & 15);
        int k = kt * 32 + ((lane >> 4) << 4) + e;
        float v = (n < NOUT && k < MEM_DIM) ? w_hh[n * MEM_DIM + k] : 0.0f;
        pWhh[j] = f2bf(v);
    }
}

// =====================================================================
// Kernel 2: per-event scatter of the non-self message columns.
//   to src: [ mem[dst] | raw_msg | cos((t-lu[src])*w+b) ]
//   to dst: [ mem[src] | raw_msg | cos((t-lu[dst])*w+b) ]
// (mem_self columns are reconstructed analytically in kernel 3)
// =====================================================================
__global__ __launch_bounds__(256) void scatter_messages(
        const float* __restrict__ memory, const float* __restrict__ raw_msg,
        const float* __restrict__ time_w, const float* __restrict__ time_b,
        const int* __restrict__ last_update,
        const int* __restrict__ src, const int* __restrict__ dst,
        const int* __restrict__ t,
        float* __restrict__ summed, float* __restrict__ counts) {
    const int e  = blockIdx.x;
    const int s  = src[e];
    const int d  = dst[e];
    const int te = t[e];
    const float trs = (float)(te - last_update[s]);
    const float trd = (float)(te - last_update[d]);

    if (threadIdx.x == 0) atomic_add_f32_dev(&counts[s], 1.0f);
    if (threadIdx.x == 1) atomic_add_f32_dev(&counts[d], 1.0f);

    for (int i = threadIdx.x; i < 2 * SCAT_W; i += 256) {
        int side  = i / SCAT_W;
        int c     = i - side * SCAT_W;
        int tgt   = side ? d : s;
        int other = side ? s : d;
        float trel = side ? trd : trs;
        float val;
        if (c < MEM_DIM) {
            val = memory[other * MEM_DIM + c];
        } else if (c < MEM_DIM + MSG_DIM) {
            val = raw_msg[e * MSG_DIM + (c - MEM_DIM)];
        } else {
            int j = c - (MEM_DIM + MSG_DIM);
            val = cosf(trel * time_w[j] + time_b[j]);
        }
        atomic_add_f32_dev(&summed[tgt * SCAT_W + c], val);
    }
}

// =====================================================================
// Kernel 3: fused mean + dual GEMM (bf16 WMMA, f32 accum) + GRU gates.
// One block = 32 nodes = 2 x 16-row M-subtiles; each B fragment load
// feeds TWO WMMAs (both subtiles), halving weight traffic from L2.
// A-matrices pre-swizzled in LDS into WMMA A-fragment layout.
// =====================================================================
__global__ __launch_bounds__(256) void gru_update(
        const float* __restrict__ memory,
        const float* __restrict__ b_ih, const float* __restrict__ b_hh,
        const float* __restrict__ summed, const float* __restrict__ counts,
        const __bf16* __restrict__ pWih, const __bf16* __restrict__ pWhh,
        float* __restrict__ out) {
    __shared__ __align__(32) __bf16 sA1[KT1][2][32][16];  // aggr, 32x480
    __shared__ __align__(32) __bf16 sA2[KT2][2][32][16];  // memory, 32x128
    __shared__ float sMem[MTILE][MEM_DIM];
    __shared__ float sGi[MTILE][GPAD];
    __shared__ float sGh[MTILE][GPAD];
    __shared__ float sCnt[MTILE];

    const int tid  = threadIdx.x;
    const int base = blockIdx.x * MTILE;

    if (tid < MTILE) sCnt[tid] = counts[base + tid];
    __syncthreads();

    for (int i = tid; i < MTILE * MEM_DIM; i += 256) {
        int r = i / MEM_DIM, c = i % MEM_DIM;
        sMem[r][c] = memory[(base + r) * MEM_DIM + c];
    }
    // aggr A-fragments (cols 0..99 = memory exactly; 100..471 = summed/count)
    for (int i = tid; i < KT1 * 2 * 512; i += 256) {
        int kt = i >> 10, r = i & 1023;
        int msub = r >> 9, q = r & 511;
        int lane = q >> 4, e = q & 15;
        int m = msub * 16 + (lane & 15), hi = lane >> 4;
        int k = kt * 32 + 16 * hi + e;
        int node = base + m;
        float v;
        if (k < MEM_DIM) {
            v = memory[node * MEM_DIM + k];
        } else if (k < K1) {
            float rc = 1.0f / fmaxf(sCnt[m], 1.0f);
            v = summed[node * SCAT_W + (k - MEM_DIM)] * rc;
        } else {
            v = 0.0f;
        }
        sA1[kt][msub][lane][e] = f2bf(v);
    }
    // memory A-fragments (padded 100 -> 128)
    for (int i = tid; i < KT2 * 2 * 512; i += 256) {
        int kt = i >> 10, r = i & 1023;
        int msub = r >> 9, q = r & 511;
        int lane = q >> 4, e = q & 15;
        int m = msub * 16 + (lane & 15), hi = lane >> 4;
        int k = kt * 32 + 16 * hi + e;
        float v = (k < MEM_DIM) ? memory[(base + m) * MEM_DIM + k] : 0.0f;
        sA2[kt][msub][lane][e] = f2bf(v);
    }
    __syncthreads();

    const int wave = tid >> 5;
    const int lane = tid & 31;
    const int hi   = lane >> 4;
    const int nloc = lane & 15;

    for (int nt = wave; nt < NTILES; nt += 8) {   // wave-uniform (EXEC all ones)
        v8f acc1a = {}, acc1b = {};
        v8f acc2a = {}, acc2b = {};
        #pragma unroll
        for (int kt = 0; kt < KT1; ++kt) {
            v16bf b  = *(const v16bf*)(pWih + ((kt * NTILES + nt) * 512 + lane * 16));
            v16bf a0 = *(const v16bf*)&sA1[kt][0][lane][0];
            v16bf a1 = *(const v16bf*)&sA1[kt][1][lane][0];
            acc1a = __builtin_amdgcn_wmma_f32_16x16x32_bf16(
                        false, a0, false, b, (short)0, acc1a, false, false);
            acc1b = __builtin_amdgcn_wmma_f32_16x16x32_bf16(
                        false, a1, false, b, (short)0, acc1b, false, false);
        }
        #pragma unroll
        for (int kt = 0; kt < KT2; ++kt) {
            v16bf b  = *(const v16bf*)(pWhh + ((kt * NTILES + nt) * 512 + lane * 16));
            v16bf a0 = *(const v16bf*)&sA2[kt][0][lane][0];
            v16bf a1 = *(const v16bf*)&sA2[kt][1][lane][0];
            acc2a = __builtin_amdgcn_wmma_f32_16x16x32_bf16(
                        false, a0, false, b, (short)0, acc2a, false, false);
            acc2b = __builtin_amdgcn_wmma_f32_16x16x32_bf16(
                        false, a1, false, b, (short)0, acc2b, false, false);
        }
        int n = nt * 16 + nloc;                  // gate column (>=300 is zero pad)
        float bih = (n < NOUT) ? b_ih[n] : 0.0f;
        float bhh = (n < NOUT) ? b_hh[n] : 0.0f;
        #pragma unroll
        for (int r = 0; r < 8; ++r) {
            int m0 = r + 8 * hi;
            int m1 = 16 + m0;
            sGi[m0][n] = acc1a[r] + bih;
            sGi[m1][n] = acc1b[r] + bih;
            sGh[m0][n] = acc2a[r] + bhh;
            sGh[m1][n] = acc2b[r] + bhh;
        }
    }
    __syncthreads();

    for (int i = tid; i < MTILE * MEM_DIM; i += 256) {
        int r = i / MEM_DIM, j = i % MEM_DIM;
        float ir = sGi[r][j],              hr = sGh[r][j];
        float iz = sGi[r][j + MEM_DIM],    hz = sGh[r][j + MEM_DIM];
        float in_ = sGi[r][j + 2*MEM_DIM], hn = sGh[r][j + 2*MEM_DIM];
        float rg = sigmoidf_(ir + hr);
        float zg = sigmoidf_(iz + hz);
        float ng = tanhf(in_ + rg * hn);
        float hprev = sMem[r][j];
        float hnew  = (1.0f - zg) * ng + zg * hprev;
        out[(base + r) * MEM_DIM + j] = (sCnt[r] > 0.0f) ? hnew : hprev;
    }
}

// =====================================================================
extern "C" void kernel_launch(void* const* d_in, const int* in_sizes, int n_in,
                              void* d_out, int out_size, void* d_ws, size_t ws_size,
                              hipStream_t stream) {
    const float* memory      = (const float*)d_in[0];
    const float* raw_msg     = (const float*)d_in[1];
    const float* time_w      = (const float*)d_in[2];
    const float* time_b      = (const float*)d_in[3];
    const float* w_ih        = (const float*)d_in[4];
    const float* w_hh        = (const float*)d_in[5];
    const float* b_ih        = (const float*)d_in[6];
    const float* b_hh        = (const float*)d_in[7];
    const int*   last_update = (const int*)d_in[8];
    const int*   src         = (const int*)d_in[9];
    const int*   dst         = (const int*)d_in[10];
    const int*   t           = (const int*)d_in[11];
    float*       out         = (float*)d_out;

    char* ws = (char*)d_ws;
    const size_t SUMMED_BYTES = (size_t)N_NODES * SCAT_W * sizeof(float); // 297.6 MB
    const size_t COUNTS_BYTES = (size_t)N_NODES * sizeof(float);
    const size_t PWIH_BYTES   = (size_t)KT1 * NTILES * 512 * sizeof(__bf16);
    float*  summed = (float*)(ws);
    float*  counts = (float*)(ws + SUMMED_BYTES);
    __bf16* pWih   = (__bf16*)(ws + SUMMED_BYTES + COUNTS_BYTES);
    __bf16* pWhh   = (__bf16*)(ws + SUMMED_BYTES + COUNTS_BYTES + PWIH_BYTES);

    // zero accumulation buffers (graph-capture safe)
    hipMemsetAsync(ws, 0, SUMMED_BYTES + COUNTS_BYTES, stream);

    const int packElems = (KT1 + KT2) * NTILES * 512;
    pack_weights<<<(packElems + 255) / 256, 256, 0, stream>>>(w_ih, w_hh, pWih, pWhh);

    scatter_messages<<<N_EVENTS, 256, 0, stream>>>(
        memory, raw_msg, time_w, time_b, last_update, src, dst, t, summed, counts);

    gru_update<<<NBLOCKS, 256, 0, stream>>>(
        memory, b_ih, b_hh, summed, counts, pWih, pWhh, out);
}
